// DecoderCell_7550552506831
// MI455X (gfx1250) — matrix operand order repaired and checked by reading, not compiled, not run on hardware
//
#include <hip/hip_runtime.h>
#include <hip/hip_bf16.h>

// DecoderCell (GRU-like) for MI455X / gfx1250, wave32 + WMMA bf16 + async LDS.
//
// Phase A: one-shot f32 -> bf16 conversion of x, hprev, c and all 9 weight
//          matrices into d_ws (memory-bound, ~11us at 23.3 TB/s).
// Phase B: three fused GEMMs, M=8192 N=1024 K=4096 on bf16 operands:
//   pass0 (r): rh = bf16(sigmoid(pre_r) * hprev)      -> ws
//   pass1 (z): z  = sigmoid(pre_z)  (f32)             -> ws
//   pass2 (h): out = z*tanh(pre_h) + (1-z)*hprev      -> d_out
// GEMM staging uses global_load_async_to_lds_b128 (ASYNCcnt), double-buffered
// so stage ks+1 streams into LDS while stage ks runs on the WMMA pipes.
//
// d_ws layout (requires 136 MB):
//   [bf16] xb(16M) hb(16M) cb(32M) | Wr Ur Cr Wz Uz Cz Wh Uh Ch (24M) | rh(16M)
//   [f32 ] z (32M)

#define B_DIM 8192
#define H_DIM 1024

typedef __attribute__((ext_vector_type(16))) __bf16 v16bf;
typedef __attribute__((ext_vector_type(8)))  __bf16 v8bf;
typedef __attribute__((ext_vector_type(8)))  float  v8f;
typedef unsigned short u16;

__device__ __forceinline__ u16 f2bf(float f) {
    unsigned u = __float_as_uint(f);
    u += 0x7FFFu + ((u >> 16) & 1u);
    return (u16)(u >> 16);
}
__device__ __forceinline__ float fast_sigmoid(float v) {
    return 1.0f / (1.0f + __expf(-v));
}
__device__ __forceinline__ float fast_tanh(float v) {
    float e = __expf(-2.0f * v);
    return (1.0f - e) / (1.0f + e);
}

// ---------------- Phase A: elementwise f32 -> bf16 (RNE) ----------------
__global__ __launch_bounds__(256) void cvt_bf16(const float* __restrict__ in,
                                                u16* __restrict__ out, long n) {
    long i = ((long)blockIdx.x * 256 + threadIdx.x) * 4;
    if (i >= n) return;
    const float4 v = *(const float4*)(in + i);
    uint2 p;
    p.x = (unsigned)f2bf(v.x) | ((unsigned)f2bf(v.y) << 16);
    p.y = (unsigned)f2bf(v.z) | ((unsigned)f2bf(v.w) << 16);
    *(uint2*)(out + i) = p;
}

// ---------------- Phase B: bf16 WMMA GEMM with async-LDS staging --------
// MODE 0: rh(bf16) = sigmoid(acc+b)*hprev ; MODE 1: z(f32) = sigmoid(acc+b)
// MODE 2: out(f32) = z*tanh(acc+b) + (1-z)*hprev
template <int MODE>
__global__ __launch_bounds__(256) void gate_gemm(
    const u16* __restrict__ A0,   // x   bf16 (B x 1024)
    const u16* __restrict__ A1,   // h/rh bf16 (B x 1024)
    const u16* __restrict__ A2,   // c   bf16 (B x 2048)
    const u16* __restrict__ W0,   // Wg bf16 (1024 x 1024)
    const u16* __restrict__ W1,   // Ug bf16 (1024 x 1024)
    const u16* __restrict__ W2,   // Cg bf16 (1024 x 2048)
    const float* __restrict__ bias,
    const float* __restrict__ hprev,
    const float* __restrict__ zgate,
    float* __restrict__ outf,
    u16*  __restrict__ outb)
{
    // K-panel 128 rows x 64 halves, row padded to 72 halves (144 B).
    __shared__ u16 Ash[2][128 * 72];
    __shared__ u16 Bsh[2][128 * 72];

    const int tid    = threadIdx.x;
    const int lane   = tid & 31;
    const int wid    = tid >> 5;
    const int wm     = wid >> 2;   // 0..1 (64 rows)
    const int wn     = wid & 3;    // 0..3 (32 cols)
    const int blockM = blockIdx.x * 128;
    const int blockN = blockIdx.y * 128;
    const int lhalf  = lane >> 4;
    const int lmod   = lane & 15;

    v8f acc[4][2];
#pragma unroll
    for (int mt = 0; mt < 4; ++mt)
#pragma unroll
        for (int nt = 0; nt < 2; ++nt)
            acc[mt][nt] = {};

    // Issue one stage: 16KB A-panel + 16KB B-panel, 8 async b128 per thread.
    auto issue = [&](int s, int b) {
        const int k0 = s * 64;
        const u16* aseg; const u16* wseg; int lda, kloc;
        if (k0 < 1024)      { aseg = A0; wseg = W0; lda = 1024; kloc = k0; }
        else if (k0 < 2048) { aseg = A1; wseg = W1; lda = 1024; kloc = k0 - 1024; }
        else                { aseg = A2; wseg = W2; lda = 2048; kloc = k0 - 2048; }
#pragma unroll
        for (int i = 0; i < 4; ++i) {
            const int idx = tid + i * 256;
            const int row = idx >> 3;   // 0..127
            const int ch  = idx & 7;    // 16B chunk within 128B row
            unsigned aoff = (unsigned)(size_t)&Ash[b][row * 72 + ch * 8];
            unsigned long long ga =
                (unsigned long long)(size_t)(aseg + (size_t)(blockM + row) * lda + kloc + ch * 8);
            asm volatile("global_load_async_to_lds_b128 %0, %1, off"
                         :: "v"(aoff), "v"(ga) : "memory");
            unsigned boff = (unsigned)(size_t)&Bsh[b][row * 72 + ch * 8];
            unsigned long long gb =
                (unsigned long long)(size_t)(wseg + (size_t)(blockN + row) * lda + kloc + ch * 8);
            asm volatile("global_load_async_to_lds_b128 %0, %1, off"
                         :: "v"(boff), "v"(gb) : "memory");
        }
    };

    auto compute = [&](int b) {
#pragma unroll
        for (int kb = 0; kb < 64; kb += 32) {
            // B frag: lane<16 -> n=lane, K kb+0..15 ; lane>=16 -> K kb+16..31
            v16bf bfrag[2];
#pragma unroll
            for (int nt = 0; nt < 2; ++nt) {
                const u16* bp = &Bsh[b][(wn * 32 + nt * 16 + lmod) * 72 + kb + lhalf * 16];
                v8bf blo = *(const v8bf*)bp;
                v8bf bhi = *(const v8bf*)(bp + 8);
                bfrag[nt] = __builtin_shufflevector(
                    blo, bhi, 0,1,2,3,4,5,6,7,8,9,10,11,12,13,14,15);
            }
            // A frag: lane<16 -> K kb+0..15 ; lane>=16 -> K kb+16..31
#pragma unroll
            for (int mt = 0; mt < 4; ++mt) {
                const u16* ap = &Ash[b][(wm * 64 + mt * 16 + lmod) * 72 + kb + lhalf * 16];
                v8bf alo = *(const v8bf*)ap;
                v8bf ahi = *(const v8bf*)(ap + 8);
                v16bf afrag = __builtin_shufflevector(
                    alo, ahi, 0,1,2,3,4,5,6,7,8,9,10,11,12,13,14,15);
#pragma unroll
                for (int nt = 0; nt < 2; ++nt) {
                    acc[mt][nt] = __builtin_amdgcn_wmma_f32_16x16x32_bf16(
                        false, afrag, false, bfrag[nt],
                        (short)0, acc[mt][nt], false, false);
                }
            }
        }
    };

    // Double-buffered pipeline over 64 K-stages. Each wave issues exactly 8
    // async loads per stage; waiting ASYNCcnt<=8 retires the older stage.
    issue(0, 0);
    for (int ks = 0; ks < 64; ++ks) {
        if (ks + 1 < 64) {
            issue(ks + 1, (ks + 1) & 1);
            asm volatile("s_wait_asynccnt 0x8" ::: "memory");
        } else {
            asm volatile("s_wait_asynccnt 0x0" ::: "memory");
        }
        __syncthreads();          // all waves' stage-ks data visible in LDS
        compute(ks & 1);
        __syncthreads();          // protect buffer before stage ks+2 overwrite
    }

    // Epilogue from the 16x16 f32 C/D layout:
    //   element (m = g + (lane>=16)*8, n = lane%16) is acc[..][..][g]
#pragma unroll
    for (int mt = 0; mt < 4; ++mt) {
#pragma unroll
        for (int nt = 0; nt < 2; ++nt) {
            const int colg = blockN + wn * 32 + nt * 16 + lmod;
            const float bv = bias[colg];
#pragma unroll
            for (int g = 0; g < 8; ++g) {
                const int rowg = blockM + wm * 64 + mt * 16 + lhalf * 8 + g;
                const size_t idx = (size_t)rowg * H_DIM + colg;
                const float v = acc[mt][nt][g] + bv;
                if (MODE == 0) {
                    outb[idx] = f2bf(fast_sigmoid(v) * hprev[idx]);
                } else if (MODE == 1) {
                    outf[idx] = fast_sigmoid(v);
                } else {
                    const float zz = zgate[idx];
                    outf[idx] = zz * fast_tanh(v) + (1.0f - zz) * hprev[idx];
                }
            }
        }
    }
}

extern "C" void kernel_launch(void* const* d_in, const int* in_sizes, int n_in,
                              void* d_out, int out_size, void* d_ws, size_t ws_size,
                              hipStream_t stream) {
    const float* x     = (const float*)d_in[0];
    const float* hprev = (const float*)d_in[1];
    const float* c     = (const float*)d_in[2];
    const float* Wh    = (const float*)d_in[3];
    const float* Wz    = (const float*)d_in[4];
    const float* Wr    = (const float*)d_in[5];
    const float* Uh    = (const float*)d_in[6];
    const float* Uz    = (const float*)d_in[7];
    const float* Ur    = (const float*)d_in[8];
    const float* Ch    = (const float*)d_in[9];
    const float* Cz    = (const float*)d_in[10];
    const float* Cr    = (const float*)d_in[11];
    const float* bh    = (const float*)d_in[12];
    const float* bz    = (const float*)d_in[13];
    const float* br    = (const float*)d_in[14];
    float* out = (float*)d_out;

    const long N_XH = (long)B_DIM * 1024;   // 8M elems
    const long N_C  = (long)B_DIM * 2048;   // 16M
    const long N_W  = 1024L * 1024;         // 1M
    const long N_CW = 1024L * 2048;         // 2M

    // ws layout (u16 units unless noted)
    u16* w16 = (u16*)d_ws;
    u16* xb  = w16;                 // 8M
    u16* hb  = xb  + N_XH;          // 8M
    u16* cb  = hb  + N_XH;          // 16M
    u16* wrb = cb  + N_C;           // 1M
    u16* urb = wrb + N_W;           // 1M
    u16* crb = urb + N_W;           // 2M
    u16* wzb = crb + N_CW;
    u16* uzb = wzb + N_W;
    u16* czb = uzb + N_W;
    u16* whb = czb + N_CW;
    u16* uhb = whb + N_W;
    u16* chb = uhb + N_W;
    u16* rhb = chb + N_CW;          // 8M (bf16 r*hprev)
    float* zf = (float*)(rhb + N_XH);   // 8M f32

    // Phase A: one-shot conversions
    auto cvt = [&](const float* src, u16* dst, long n) {
        cvt_bf16<<<dim3((unsigned)(n / 1024)), dim3(256), 0, stream>>>(src, dst, n);
    };
    cvt(x, xb, N_XH);  cvt(hprev, hb, N_XH);  cvt(c, cb, N_C);
    cvt(Wr, wrb, N_W); cvt(Ur, urb, N_W);     cvt(Cr, crb, N_CW);
    cvt(Wz, wzb, N_W); cvt(Uz, uzb, N_W);     cvt(Cz, czb, N_CW);
    cvt(Wh, whb, N_W); cvt(Uh, uhb, N_W);     cvt(Ch, chb, N_CW);

    // Phase B: three WMMA GEMM passes
    dim3 grid(B_DIM / 128, H_DIM / 128);
    dim3 block(256);
    gate_gemm<0><<<grid, block, 0, stream>>>(xb, hb,  cb, wrb, urb, crb, br, hprev, nullptr, nullptr, rhb);
    gate_gemm<1><<<grid, block, 0, stream>>>(xb, hb,  cb, wzb, uzb, czb, bz, nullptr, nullptr, zf,  nullptr);
    gate_gemm<2><<<grid, block, 0, stream>>>(xb, rhb, cb, whb, uhb, chb, bh, hprev, zf,      out,  nullptr);
}
